// BiAffine_9569187135520
// MI455X (gfx1250) — matrix operand order repaired and checked by reading, not compiled, not run on hardware
//
#include <hip/hip_runtime.h>
#include <hip/hip_bf16.h>

typedef __bf16 bf16;
typedef __attribute__((ext_vector_type(8)))  __bf16 v8bf;
typedef __attribute__((ext_vector_type(16))) __bf16 v16bf;
typedef __attribute__((ext_vector_type(8)))  float  v8f;

__device__ __forceinline__ v16bf cat8(v8bf lo, v8bf hi) {
    return __builtin_shufflevector(lo, hi, 0,1,2,3,4,5,6,7,8,9,10,11,12,13,14,15);
}

// CDNA5 async global->LDS copy, 16B per lane, ASYNCcnt-tracked (ISA §15.18 op 98).
__device__ __forceinline__ void async_cp16(unsigned lds_off, const bf16* gptr) {
    asm volatile("global_load_async_to_lds_b128 %0, %1, off"
                 :: "v"(lds_off), "v"(gptr) : "memory");
}
__device__ __forceinline__ void wait_async0() {
    asm volatile("s_wait_asynccnt 0" ::: "memory");
}

// ---------------------------------------------------------------------------
// f32 -> bf16 conversion (grid-stride)
// ---------------------------------------------------------------------------
__global__ __launch_bounds__(256) void cvt_f32_bf16(const float* __restrict__ x,
                                                    bf16* __restrict__ y, size_t n) {
    size_t i = (size_t)blockIdx.x * blockDim.x + threadIdx.x;
    size_t stride = (size_t)gridDim.x * blockDim.x;
    for (; i < n; i += stride) y[i] = (bf16)x[i];
}

// ---------------------------------------------------------------------------
// Fused transpose + convert: x (f32, [2048][1024] per batch) -> xt (bf16,
// [1024][2048] per batch). 32x32 LDS tile, coalesced both sides.
// ---------------------------------------------------------------------------
__global__ __launch_bounds__(256) void transpose_cvt(const float* __restrict__ x,
                                                     bf16* __restrict__ xt) {
    __shared__ float tile[32][33];
    const size_t bo = (size_t)blockIdx.z * 2048 * 1024;
    const int h0 = blockIdx.x * 32;
    const int n0 = blockIdx.y * 32;
    const int tx = threadIdx.x & 31;
    const int ty = threadIdx.x >> 5;   // 8 rows per pass
    #pragma unroll
    for (int i = 0; i < 32; i += 8)
        tile[ty + i][tx] = x[bo + (size_t)(n0 + ty + i) * 1024 + h0 + tx];
    __syncthreads();
    #pragma unroll
    for (int i = 0; i < 32; i += 8)
        xt[bo + (size_t)(h0 + ty + i) * 2048 + n0 + tx] = (bf16)tile[tx][ty + i];
}

// ---------------------------------------------------------------------------
// NT GEMM: C[m,n] = sum_k A[m,k] * B[n,k]   (A: MxK row-major, B: NxK row-major)
// Block: 256 thr = 8 waves (4x2); C tile 256(M) x 128(N); wave tile 64x64;
// K step 32. Tiles stream global->LDS via async-to-LDS copies, double-buffered.
// Per wave per k-step: 16 ds_load_b128 -> 16 v_wmma (1:1).
// ---------------------------------------------------------------------------
template <typename OutT>
__global__ __launch_bounds__(256) void gemm_nt_wmma(
    const bf16* __restrict__ A, int lda,
    const bf16* __restrict__ B, int ldb,
    OutT* __restrict__ C, int ldc, int K,
    long long sA, long long sB, long long sC)
{
    A += (long long)blockIdx.z * sA;
    B += (long long)blockIdx.z * sB;
    C += (long long)blockIdx.z * sC;
    const int n0 = blockIdx.x * 128;
    const int m0 = blockIdx.y * 256;

    __shared__ bf16 As[2][256][40];   // 80B pitch: every 16B frag read stays aligned
    __shared__ bf16 Bs[2][128][40];

    const int t    = threadIdx.x;
    const int lane = t & 31;
    const int wave = t >> 5;
    const int wx   = wave & 1;        // 2 waves along N (64 each)
    const int wy   = wave >> 1;       // 4 waves along M (64 each)
    const int r    = lane & 15;
    const int hh   = lane >> 4;

    v8f acc[4][4];
    #pragma unroll
    for (int i = 0; i < 4; ++i)
        #pragma unroll
        for (int j = 0; j < 4; ++j)
            #pragma unroll
            for (int q = 0; q < 8; ++q) acc[i][j][q] = 0.0f;

    // staging map: 16-half (32B... 16B) chunks; A tile = 512 chunks, B tile = 256
    const int ar0 = t >> 1;            // A chunk 0 row  (0..127)
    const int ar1 = 128 + (t >> 1);    // A chunk 1 row  (128..255)
    const int sg  = t & 1;             // segment within 32-half row

    auto issue = [&](int kk, int buf) {
        const bf16* ga0 = A + (size_t)(m0 + ar0) * lda + kk + sg * 16;
        async_cp16((unsigned)(uintptr_t)&As[buf][ar0][sg * 16], ga0);
        const bf16* ga1 = A + (size_t)(m0 + ar1) * lda + kk + sg * 16;
        async_cp16((unsigned)(uintptr_t)&As[buf][ar1][sg * 16], ga1);
        const bf16* gb = B + (size_t)(n0 + ar0) * ldb + kk + sg * 16;
        async_cp16((unsigned)(uintptr_t)&Bs[buf][ar0][sg * 16], gb);
    };

    const int nk = K >> 5;
    issue(0, 0);
    for (int i = 0; i < nk; ++i) {
        wait_async0();
        __syncthreads();                       // buf[i&1] fully resident for all waves
        if (i + 1 < nk) issue((i + 1) << 5, (i + 1) & 1);
        const int buf = i & 1;

        v16bf af[4], bfr[4];
        #pragma unroll
        for (int mi = 0; mi < 4; ++mi) {
            const bf16* ap = &As[buf][wy * 64 + mi * 16 + r][hh * 8];
            af[mi] = cat8(*(const v8bf*)ap, *(const v8bf*)(ap + 16));
        }
        #pragma unroll
        for (int ni = 0; ni < 4; ++ni) {
            const bf16* bp = &Bs[buf][wx * 64 + ni * 16 + (lane & 15)][(lane >> 4) * 8];
            bfr[ni] = cat8(*(const v8bf*)bp, *(const v8bf*)(bp + 16));
        }
        #pragma unroll
        for (int mi = 0; mi < 4; ++mi)
            #pragma unroll
            for (int ni = 0; ni < 4; ++ni)
                acc[mi][ni] = __builtin_amdgcn_wmma_f32_16x16x32_bf16(
                    false, af[mi], false, bfr[ni], (short)0, acc[mi][ni], false, false);
    }

    // epilogue: C-layout — lane holds col (lane&15), rows (lane>>4)*8 + q
    #pragma unroll
    for (int mi = 0; mi < 4; ++mi)
        #pragma unroll
        for (int ni = 0; ni < 4; ++ni) {
            const int cc = n0 + wx * 64 + ni * 16 + (lane & 15);
            const int rb = m0 + wy * 64 + mi * 16 + (lane >> 4) * 8;
            #pragma unroll
            for (int q = 0; q < 8; ++q)
                C[(size_t)(rb + q) * ldc + cc] = (OutT)acc[mi][ni][q];
        }
}

// ---------------------------------------------------------------------------
// In-place row softmax over 2048 bf16 columns. One 256-thread block per row.
// ---------------------------------------------------------------------------
__global__ __launch_bounds__(256) void softmax_rows2048(bf16* __restrict__ s) {
    bf16* p = s + (size_t)blockIdx.x * 2048;
    const int t    = threadIdx.x;
    const int lane = t & 31;
    const int wave = t >> 5;

    float v[8];
    float m = -3.4e38f;
    #pragma unroll
    for (int i = 0; i < 8; ++i) {
        v[i] = (float)p[t + i * 256];
        m = fmaxf(m, v[i]);
    }
    #pragma unroll
    for (int off = 16; off > 0; off >>= 1) m = fmaxf(m, __shfl_xor(m, off, 32));

    __shared__ float red[8];
    if (lane == 0) red[wave] = m;
    __syncthreads();
    if (wave == 0) {
        float mm = (lane < 8) ? red[lane] : -3.4e38f;
        #pragma unroll
        for (int off = 16; off > 0; off >>= 1) mm = fmaxf(mm, __shfl_xor(mm, off, 32));
        if (lane == 0) red[0] = mm;
    }
    __syncthreads();
    m = red[0];
    __syncthreads();

    float sum = 0.0f;
    #pragma unroll
    for (int i = 0; i < 8; ++i) {
        v[i] = __expf(v[i] - m);
        sum += v[i];
    }
    #pragma unroll
    for (int off = 16; off > 0; off >>= 1) sum += __shfl_xor(sum, off, 32);
    if (lane == 0) red[wave] = sum;
    __syncthreads();
    if (wave == 0) {
        float ss = (lane < 8) ? red[lane] : 0.0f;
        #pragma unroll
        for (int off = 16; off > 0; off >>= 1) ss += __shfl_xor(ss, off, 32);
        if (lane == 0) red[0] = ss;
    }
    __syncthreads();
    const float inv = 1.0f / red[0];
    #pragma unroll
    for (int i = 0; i < 8; ++i) p[t + i * 256] = (bf16)(v[i] * inv);
}

// ---------------------------------------------------------------------------
extern "C" void kernel_launch(void* const* d_in, const int* in_sizes, int n_in,
                              void* d_out, int out_size, void* d_ws, size_t ws_size,
                              hipStream_t stream) {
    (void)in_sizes; (void)n_in; (void)out_size; (void)ws_size;
    const float* S1 = (const float*)d_in[0];
    const float* S2 = (const float*)d_in[1];
    const float* W1 = (const float*)d_in[2];
    const float* W2 = (const float*)d_in[3];

    const int Bz = 8, N = 2048, H = 1024;
    const size_t eS = (size_t)Bz * N * H;   // 16,777,216
    const size_t eW = (size_t)H * H;        //  1,048,576

    // workspace carve (bf16): S1b|S2b|W1b|W2b|Pj|SC|S1t|S2t  ~= 239 MB
    bf16* S1b = (bf16*)d_ws;
    bf16* S2b = S1b + eS;
    bf16* W1b = S2b + eS;
    bf16* W2b = W1b + eW;
    bf16* Pj  = W2b + eW;                   // projection, eS
    bf16* SC  = Pj  + eS;                   // scores/probs, Bz*N*N
    bf16* S1t = SC  + (size_t)Bz * N * N;   // S1 transposed per batch [H][N]
    bf16* S2t = S1t + eS;                   // S2 transposed per batch [H][N]
    float* O1 = (float*)d_out;
    float* O2 = O1 + eS;

    const long long sNH = (long long)N * H;
    const long long sNN = (long long)N * N;

    cvt_f32_bf16<<<4096, 256, 0, stream>>>(S1, S1b, eS);
    cvt_f32_bf16<<<4096, 256, 0, stream>>>(S2, S2b, eS);
    cvt_f32_bf16<<<1024, 256, 0, stream>>>(W1, W1b, eW);
    cvt_f32_bf16<<<1024, 256, 0, stream>>>(W2, W2b, eW);
    transpose_cvt<<<dim3(H / 32, N / 32, Bz), 256, 0, stream>>>(S1, S1t);
    transpose_cvt<<<dim3(H / 32, N / 32, Bz), 256, 0, stream>>>(S2, S2t);

    // ---- direction 1: S1_out = softmax(S1 (S2 W1^T)^T) S2 ----
    gemm_nt_wmma<bf16><<<dim3(H / 128, (Bz * N) / 256, 1), 256, 0, stream>>>(
        S2b, H, W1b, H, Pj, H, H, 0, 0, 0);
    gemm_nt_wmma<bf16><<<dim3(N / 128, N / 256, Bz), 256, 0, stream>>>(
        S1b, H, Pj, H, SC, N, H, sNH, sNH, sNN);
    softmax_rows2048<<<Bz * N, 256, 0, stream>>>(SC);
    // O1[b][m,h] = sum_n P[m,n] * S2t[h,n]  -> NT GEMM against transposed V
    gemm_nt_wmma<float><<<dim3(H / 128, N / 256, Bz), 256, 0, stream>>>(
        SC, N, S2t, N, O1, H, N, sNN, sNH, sNH);

    // ---- direction 2: S2_out = softmax(S2 (S1 W2^T)^T) S1 ----
    gemm_nt_wmma<bf16><<<dim3(H / 128, (Bz * N) / 256, 1), 256, 0, stream>>>(
        S1b, H, W2b, H, Pj, H, H, 0, 0, 0);
    gemm_nt_wmma<bf16><<<dim3(N / 128, N / 256, Bz), 256, 0, stream>>>(
        S2b, H, Pj, H, SC, N, H, sNH, sNH, sNN);
    softmax_rows2048<<<Bz * N, 256, 0, stream>>>(SC);
    gemm_nt_wmma<float><<<dim3(H / 128, N / 256, Bz), 256, 0, stream>>>(
        SC, N, S1t, N, O2, H, N, sNN, sNH, sNH);
}
